// Net_12481174962490
// MI455X (gfx1250) — matrix operand order repaired and checked by reading, not compiled, not run on hardware
//
#include <hip/hip_runtime.h>
#include <hip/hip_bf16.h>
#include <math.h>

typedef __attribute__((ext_vector_type(16))) _Float16 v16h;
typedef __attribute__((ext_vector_type(8)))  _Float16 v8h;
typedef __attribute__((ext_vector_type(8)))  float    v8f;
typedef __attribute__((ext_vector_type(4)))  unsigned int u32x4;
typedef __attribute__((ext_vector_type(8)))  int i32x8;
typedef __attribute__((ext_vector_type(4)))  int i32x4;

// ---------------------------------------------------------------------------
// WMMA fragment layouts used below (wave32, v_wmma_f32_16x16x32_f16):
//  A 16x32 f16: lane(0-15) row M=lane, halves j=0..7 -> K=j+8*hi,
//               j=8..15 -> K=16+(j-8)+8*hi   (hi = lane>=16)
//  B 32x16 f16: col N=lane%16, halves j=0..7 -> K=j+8*hi,
//               j=8..15 -> K=16+(j-8)+8*hi   (same K order as A)
//  C/D 16x16 f32: lanes 0-15 col=lane M=r; lanes 16-31 col=lane-16 M=r+8
// ---------------------------------------------------------------------------

// ======================= big-Cout conv: 64M x 64N block =====================
// 4 waves share one LDS-staged activation tile; weight tile arrives via the
// Tensor Data Mover (tensor_load_to_lds) with LDS row padding 32->40 halves.
template<int KH, int KW>
__global__ __launch_bounds__(128) void conv_wmma_big_k(
    const float* __restrict__ x, const _Float16* __restrict__ wt,
    const float* __restrict__ bias, float* __restrict__ y,
    const float* __restrict__ residual,
    int B, int Cin, int H, int W, int Cout, int pad, int Kpad, int Mpad, int act)
{
    constexpr int KK = KH * KW;
    constexpr int RS = 40;                 // LDS row stride in halves (32 + 8 pad)
    __shared__ __align__(16) _Float16 smw[64 * RS];   // weight tile (TDM dest, offset 0)
    __shared__ __align__(16) _Float16 smb[64 * RS];   // activation tile

    const int tid  = threadIdx.x;
    const int lane = tid & 31, wave = tid >> 5;
    const int hi   = lane >> 4, lr = lane & 15;
    const int HW = H * W, Ntot = B * HW, Kreal = Cin * KK;
    const int nbase = blockIdx.x * 64;
    const int mb    = blockIdx.y;

    // gather-thread decode: thread t owns column gc = t/2, K-half gh = t&1
    const int gc = tid >> 1, gh = tid & 1;
    const int gn = nbase + gc;
    const bool gOk = gn < Ntot;
    const int gnn = gOk ? gn : 0;
    const int gb = gnn / HW; const int gsp = gnn - gb * HW;
    const int goh = gsp / W, gow = gsp - goh * W;
    const size_t gibase = (size_t)gb * Cin * HW;

    // compute-lane decode of the 4 output sub-tile columns
    int bImg[4], oH[4], oW[4]; bool nOk[4];
    #pragma unroll
    for (int s = 0; s < 4; ++s) {
        int n = nbase + s * 16 + lr;
        nOk[s] = n < Ntot;
        int nn = nOk[s] ? n : 0;
        int b = nn / HW; int sp = nn - b * HW;
        bImg[s] = b; oH[s] = sp / W; oW[s] = sp - oH[s] * W;
    }

    v8f acc[4] = {};
    const int arow = wave * 16 + lr;

    for (int kb = 0; kb < Kpad; kb += 32) {
        // ---- TDM: DMA 64x32 f16 weight tile -> LDS (wave 0 issues) ----
        if (wave == 0) {
            unsigned long long ga =
                (unsigned long long)(uintptr_t)(wt + ((size_t)mb * 64) * Kpad + kb);
            int Krem = Kpad - kb;                 // remaining extent from tile start
            int Mrem = Mpad - mb * 64; if (Mrem < 0) Mrem = 0;
            u32x4 g0;
            g0[0] = 1u;                                         // count=1, user mode
            g0[1] = 0u;                                         // lds_addr: smw at 0
            g0[2] = (unsigned)ga;                               // global_addr lo
            g0[3] = (unsigned)((ga >> 32) & 0x01FFFFFFull) | 0x80000000u; // hi|type=2
            i32x8 g1;
            g1[0] = (int)((1u << 16)        // data_size = 2 bytes
                   | (1u << 20)             // pad_enable
                   | (3u << 22)             // pad_interval: every 16 dwords
                   | (3u << 25));           // pad_amount: 4 dwords -> row stride 40h
            g1[1] = (int)((unsigned)(Krem & 0xFFFF) << 16);     // tensor_dim0 lo
            g1[2] = (int)(((unsigned)Krem >> 16)
                   | ((unsigned)(Mrem & 0xFFFF) << 16));        // dim0 hi | dim1 lo
            g1[3] = (int)(32u << 16);                           // dim1 hi | tile_dim0=32
            g1[4] = 64;                                         // tile_dim1=64, dim2=0
            g1[5] = Kpad;                                       // dim0_stride lo
            g1[6] = 0;                                          // stride hi | dim1_stride
            g1[7] = 0;
            i32x4 gz4 = {};
            i32x8 gz8 = {};
            __builtin_amdgcn_tensor_load_to_lds(g0, g1, gz4, gz4, gz8, 0);
        }

        // ---- cooperative im2col gather: 64x32 activations f32 -> f16 LDS ----
        {
            v8h lo8, hi8;
            #pragma unroll
            for (int j = 0; j < 16; ++j) {
                int k = kb + gh * 16 + j;
                float val = 0.f;
                if (gOk && k < Kreal) {
                    int ci = k / KK;
                    int r  = k - ci * KK;
                    int ih = goh + r / KW - pad;
                    int iw = gow + (r - (r / KW) * KW) - pad;
                    if ((unsigned)ih < (unsigned)H && (unsigned)iw < (unsigned)W)
                        val = x[gibase + (size_t)ci * HW + ih * W + iw];
                }
                if (j < 8) lo8[j] = (_Float16)val; else hi8[j - 8] = (_Float16)val;
            }
            *(v8h*)&smb[gc * RS + gh * 16]     = lo8;
            *(v8h*)&smb[gc * RS + gh * 16 + 8] = hi8;
        }

        if (wave == 0) __builtin_amdgcn_s_wait_tensorcnt(0);
        __syncthreads();

        // ---- fragments from LDS, 4 WMMAs per wave ----
        v16h a;
        {
            v8h lo = *(const v8h*)&smw[arow * RS + hi * 8];
            v8h hh = *(const v8h*)&smw[arow * RS + 16 + hi * 8];
            #pragma unroll
            for (int j = 0; j < 8; ++j) { a[j] = lo[j]; a[j + 8] = hh[j]; }
        }
        #pragma unroll
        for (int s = 0; s < 4; ++s) {
            v16h bf;
            v8h lo = *(const v8h*)&smb[(s * 16 + lr) * RS + hi * 8];
            v8h hh = *(const v8h*)&smb[(s * 16 + lr) * RS + 16 + hi * 8];
            #pragma unroll
            for (int j = 0; j < 8; ++j) { bf[j] = lo[j]; bf[j + 8] = hh[j]; }
            acc[s] = __builtin_amdgcn_wmma_f32_16x16x32_f16(
                false, a, false, bf, (short)0, acc[s], false, false);
        }
        __syncthreads();
    }

    // ---- epilogue: bias + optional residual + activation ----
    #pragma unroll
    for (int s = 0; s < 4; ++s) {
        if (!nOk[s]) continue;
        #pragma unroll
        for (int r = 0; r < 8; ++r) {
            int m = mb * 64 + wave * 16 + r + hi * 8;
            if (m >= Cout) continue;
            float v = acc[s][r] + bias[m];
            size_t oidx = ((size_t)bImg[s] * Cout + m) * HW + oH[s] * W + oW[s];
            if (residual) v += residual[oidx];
            if (act == 1)      v = fmaxf(v, 0.f);
            else if (act == 2) v = 1.f / (1.f + __expf(-v));
            y[oidx] = v;
        }
    }
}

// ======================= small-Cout conv: wave-private 16M x 64N ============
template<int KH, int KW>
__global__ __launch_bounds__(128) void conv_wmma_k(
    const float* __restrict__ x, const _Float16* __restrict__ wt,
    const float* __restrict__ bias, float* __restrict__ y,
    const float* __restrict__ residual,
    int B, int Cin, int H, int W, int Cout, int pad, int Kpad, int act)
{
    constexpr int KK = KH * KW;
    const int lane  = threadIdx.x & 31;
    const int wave  = threadIdx.x >> 5;
    const int hi    = lane >> 4;
    const int lr    = lane & 15;
    const int HW    = H * W;
    const int Ntot  = B * HW;
    const int Kreal = Cin * KK;
    const int nbase = (blockIdx.x * 4 + wave) * 64;
    const int mrow0 = blockIdx.y * 16;

    int bImg[4], oH[4], oW[4];
    bool nOk[4];
    #pragma unroll
    for (int s = 0; s < 4; ++s) {
        int n  = nbase + s * 16 + lr;
        nOk[s] = (n < Ntot);
        int nn = nOk[s] ? n : 0;
        int b  = nn / HW;
        int sp = nn - b * HW;
        bImg[s] = b;
        oH[s]   = sp / W;
        oW[s]   = sp - oH[s] * W;
    }

    v8f acc[4] = {};
    const _Float16* wrow = wt + (size_t)(mrow0 + lr) * Kpad + hi * 8;

    for (int kb = 0; kb < Kpad; kb += 32) {
        v16h a;
        {
            v8h lo = *(const v8h*)(wrow + kb);
            v8h hh = *(const v8h*)(wrow + kb + 16);
            #pragma unroll
            for (int j = 0; j < 8; ++j) { a[j] = lo[j]; a[j + 8] = hh[j]; }
        }
        const int kcol = kb + hi * 8;

        #pragma unroll
        for (int s = 0; s < 4; ++s) {
            const size_t ibase = (size_t)bImg[s] * Cin * HW;
            v16h bf;
            #pragma unroll
            for (int j = 0; j < 16; ++j) {
                int   k   = kcol + j + (j >= 8 ? 8 : 0);
                float val = 0.f;
                if (nOk[s] && k < Kreal) {
                    int ci = k / KK;
                    int r  = k - ci * KK;
                    int ih = oH[s] + r / KW - pad;
                    int iw = oW[s] + (r - (r / KW) * KW) - pad;
                    if ((unsigned)ih < (unsigned)H && (unsigned)iw < (unsigned)W)
                        val = x[ibase + (size_t)ci * HW + ih * W + iw];
                }
                bf[j] = (_Float16)val;
            }
            acc[s] = __builtin_amdgcn_wmma_f32_16x16x32_f16(
                false, a, false, bf, (short)0, acc[s], false, false);
        }
    }

    #pragma unroll
    for (int s = 0; s < 4; ++s) {
        if (!nOk[s]) continue;
        #pragma unroll
        for (int r = 0; r < 8; ++r) {
            int m = mrow0 + r + hi * 8;
            if (m >= Cout) continue;
            float v = acc[s][r] + bias[m];
            size_t oidx = ((size_t)bImg[s] * Cout + m) * HW + oH[s] * W + oW[s];
            if (residual) v += residual[oidx];
            if (act == 1)      v = fmaxf(v, 0.f);
            else if (act == 2) v = 1.f / (1.f + __expf(-v));
            y[oidx] = v;
        }
    }
}

// ---- pack fp32 OIHW weights into zero-padded f16 [Mpad][Kpad] -------------
__global__ void pack_w_f16(const float* __restrict__ w, _Float16* __restrict__ dst,
                           int Cout, int K, int Kpad, int Mpad)
{
    int i = blockIdx.x * blockDim.x + threadIdx.x;
    int tot = Mpad * Kpad;
    if (i >= tot) return;
    int m = i / Kpad, k = i - m * Kpad;
    float v = (m < Cout && k < K) ? w[(size_t)m * K + k] : 0.f;
    dst[i] = (_Float16)v;
}

// ---- 2x2 maxpool (optional fused ReLU) ------------------------------------
__global__ void maxpool2_k(const float* __restrict__ x, float* __restrict__ y,
                           int BC, int H, int W, int relu)
{
    int OH = H >> 1, OW = W >> 1;
    size_t total = (size_t)BC * OH * OW;
    for (size_t i = (size_t)blockIdx.x * blockDim.x + threadIdx.x; i < total;
         i += (size_t)gridDim.x * blockDim.x) {
        int bc = (int)(i / ((size_t)OH * OW));
        int sp = (int)(i - (size_t)bc * OH * OW);
        int oh = sp / OW, ow = sp - oh * OW;
        const float* p = x + ((size_t)bc * H + 2 * oh) * W + 2 * ow;
        float m = fmaxf(fmaxf(p[0], p[1]), fmaxf(p[W], p[W + 1]));
        if (relu) m = fmaxf(m, 0.f);
        y[i] = m;
    }
}

// ---- training-mode BatchNorm: per-channel stats then apply ---------------
__global__ void bn_stats_k(const float* __restrict__ x, float* __restrict__ stats,
                           int Bn, int C, int HW)
{
    __shared__ float ss[256], sq[256];
    int c = blockIdx.x;
    int Nper = Bn * HW;
    float s = 0.f, q = 0.f;
    for (int i = threadIdx.x; i < Nper; i += blockDim.x) {
        int b = i / HW; int sp = i - b * HW;
        float v = x[((size_t)b * C + c) * HW + sp];
        s += v; q += v * v;
    }
    ss[threadIdx.x] = s; sq[threadIdx.x] = q;
    __syncthreads();
    for (int o = 128; o > 0; o >>= 1) {
        if (threadIdx.x < o) {
            ss[threadIdx.x] += ss[threadIdx.x + o];
            sq[threadIdx.x] += sq[threadIdx.x + o];
        }
        __syncthreads();
    }
    if (threadIdx.x == 0) {
        float mean = ss[0] / (float)Nper;
        float var  = sq[0] / (float)Nper - mean * mean;   // biased, matches jnp.var
        stats[2 * c] = mean; stats[2 * c + 1] = var;
    }
}

__global__ void bn_apply_k(float* __restrict__ x, const float* __restrict__ stats,
                           const float* __restrict__ g, const float* __restrict__ b,
                           int C, int HW, size_t total, int relu)
{
    for (size_t i = (size_t)blockIdx.x * blockDim.x + threadIdx.x; i < total;
         i += (size_t)gridDim.x * blockDim.x) {
        int c = (int)((i / (size_t)HW) % (size_t)C);
        float mean = stats[2 * c], var = stats[2 * c + 1];
        float v = (x[i] - mean) * rsqrtf(var + 1e-5f) * g[c] + b[c];
        x[i] = relu ? fmaxf(v, 0.f) : v;
    }
}

// ---- nearest resize [8,1,64,64] -> [8,1,9,9] ------------------------------
__global__ void resize9_k(const float* __restrict__ x, float* __restrict__ y)
{
    int i = blockIdx.x * blockDim.x + threadIdx.x;
    if (i >= 8 * 81) return;
    int b = i / 81, r = (i % 81) / 9, c = i % 9;
    int ih = (r * 64) / 9, iw = (c * 64) / 9;
    y[i] = x[b * 4096 + ih * 64 + iw];
}

// ---- change_blur broadcast (81 ch) + concat with [8,100,128,128] ----------
__global__ void concat_blur_k(const float* __restrict__ nine,  // [8,81]
                              const float* __restrict__ feat,  // [8,100,128,128]
                              float* __restrict__ out)         // [8,181,128,128]
{
    size_t total = (size_t)8 * 181 * 16384;
    for (size_t i = (size_t)blockIdx.x * blockDim.x + threadIdx.x; i < total;
         i += (size_t)gridDim.x * blockDim.x) {
        int b  = (int)(i / ((size_t)181 * 16384));
        int c  = (int)((i / 16384) % 181);
        int sp = (int)(i & 16383);
        out[i] = (c < 81) ? nine[b * 81 + c]
                          : feat[((size_t)b * 100 + (c - 81)) * 16384 + sp];
    }
}

// ---- pixel shuffle r=2: [8,16,128,128] -> [8,4,256,256] -------------------
__global__ void pixel_shuffle2_k(const float* __restrict__ x, float* __restrict__ y)
{
    size_t total = (size_t)8 * 4 * 65536;
    for (size_t i = (size_t)blockIdx.x * blockDim.x + threadIdx.x; i < total;
         i += (size_t)gridDim.x * blockDim.x) {
        int b  = (int)(i / ((size_t)4 * 65536));
        int c  = (int)((i / 65536) % 4);
        int sp = (int)(i & 65535);
        int h = sp >> 8, w = sp & 255;
        int cin = c * 4 + (h & 1) * 2 + (w & 1);
        y[i] = x[((size_t)b * 16 + cin) * 16384 + (h >> 1) * 128 + (w >> 1)];
    }
}

// ---------------------------------------------------------------------------
extern "C" void kernel_launch(void* const* d_in, const int* in_sizes, int n_in,
                              void* d_out, int out_size, void* d_ws, size_t ws_size,
                              hipStream_t stream)
{
    (void)in_sizes; (void)n_in; (void)out_size; (void)ws_size;
    float* ws = (float*)d_ws;
    const float* x = (const float*)d_in[0];

    // Arena (float-element offsets), liveness-scheduled. Peak ~250.5 MB.
    float* A  = ws;               // conv1 out   [8,16,256,256]
    float* Bb = ws + 8500000;     // pooled1/bn1 [8,16,128,128]
    float* C  = ws + 24000000;    // conv2 out   [8,100,128,128] (persist->concat)
    float* D  = ws + 37200000;    // pooled2     [8,100,64,64]
    float* E  = ws + 40500000;    // conv3 out   [8,32,64,64]
    float* F  = ws + 41600000;    // conv4 out   [8,9,64,64]
    float* G  = ws + 41900000;    // conv5 out   [8,1,64,64]
    float* Hh = ws + 41950000;    // resized     [8,81]
    float* I  = ws;               // concat      [8,181,128,128] (reuse A/Bb)
    float* J  = ws + 24000000;    // res1c1      [8,256,128,128] (reuse C..H)
    float* K  = ws + 57600000;    // res1c2/bn2  [8,32,128,128]
    float* L  = ws + 24000000;    // out32       [8,32,128,128]  (reuse J)
    float* M  = ws + 28200000;    // res2c1      [8,128,128,128]
    float* N  = ws;               // res2c2/bn2  [8,64,128,128]  (reuse I)
    float* O  = ws + 8400000;     // out64       [8,64,128,128]
    float* P  = ws + 17000000;    // up1         [8,30,128,128]
    float* Q  = ws;               // up2         [8,16,128,128]
    float* R  = ws + 2100000;     // shuffled    [8,4,256,256]
    float* S  = ws + 4200000;     // up3         [8,8,256,256]
    _Float16* WT = (_Float16*)(ws + 61800000);   // f16 weight arena (~713K halves)
    float* STATS = ws + 62600000;                // BN mean/var (<=512 floats)

    // conv layers: {weight input idx, Cout, Cin, KH, KW}
    static const int LW[15][5] = {
        { 1,  16,   1, 3, 3},  // 0  conv1
        { 5, 100,  16, 5, 5},  // 1  conv2
        { 7,  32, 100, 3, 3},  // 2  conv3
        { 9,   9,  32, 3, 3},  // 3  conv4
        {11,   1,   9, 1, 1},  // 4  conv5
        {13, 256, 181, 3, 3},  // 5  res1_conv1
        {17,  32, 256, 3, 3},  // 6  res1_conv2
        {21,  32, 181, 1, 1},  // 7  conv1x1_1
        {23, 128,  32, 3, 3},  // 8  res2_conv1
        {27,  64, 128, 3, 3},  // 9  res2_conv2
        {31,  64,  32, 1, 1},  // 10 conv1x1_2
        {33,  30,  64, 3, 3},  // 11 up_conv1
        {35,  16,  30, 1, 1},  // 12 up_conv2
        {37,   8,   4, 3, 3},  // 13 up_conv3
        {39,   1,   8, 1, 1},  // 14 up_conv4
    };

    _Float16* wptr[15];
    {
        size_t off = 0;
        for (int l = 0; l < 15; ++l) {
            int Cout = LW[l][1], Cin = LW[l][2], KH = LW[l][3], KW = LW[l][4];
            int Kr = Cin * KH * KW, Kp = (Kr + 31) & ~31, Mp = (Cout + 15) & ~15;
            wptr[l] = WT + off;
            int tot = Mp * Kp;
            pack_w_f16<<<(tot + 255) / 256, 256, 0, stream>>>(
                (const float*)d_in[LW[l][0]], wptr[l], Cout, Kr, Kp, Mp);
            off += (size_t)tot;
        }
    }

    auto conv = [&](const float* in, int l, const float* bias, float* out,
                    const float* resid, int H, int W, int act) {
        int Cout = LW[l][1], Cin = LW[l][2], KH = LW[l][3], KW = LW[l][4];
        int Kr = Cin * KH * KW, Kp = (Kr + 31) & ~31, Mp = (Cout + 15) & ~15;
        int pad = (KH == 3) ? 1 : (KH == 5) ? 2 : 0;
        if (Mp >= 64) {  // LDS-shared 64Mx64N block with TDM weight tiles
            dim3 g((unsigned)((8 * H * W + 63) / 64), (unsigned)((Mp + 63) / 64));
            if (KH == 3)
                conv_wmma_big_k<3,3><<<g, 128, 0, stream>>>(in, wptr[l], bias, out,
                    resid, 8, Cin, H, W, Cout, pad, Kp, Mp, act);
            else if (KH == 5)
                conv_wmma_big_k<5,5><<<g, 128, 0, stream>>>(in, wptr[l], bias, out,
                    resid, 8, Cin, H, W, Cout, pad, Kp, Mp, act);
            else
                conv_wmma_big_k<1,1><<<g, 128, 0, stream>>>(in, wptr[l], bias, out,
                    resid, 8, Cin, H, W, Cout, pad, Kp, Mp, act);
        } else {         // wave-private 16Mx64N
            dim3 g((unsigned)((8 * H * W + 255) / 256), (unsigned)(Mp / 16));
            if (KH == 3)
                conv_wmma_k<3,3><<<g, 128, 0, stream>>>(in, wptr[l], bias, out,
                    resid, 8, Cin, H, W, Cout, pad, Kp, act);
            else if (KH == 5)
                conv_wmma_k<5,5><<<g, 128, 0, stream>>>(in, wptr[l], bias, out,
                    resid, 8, Cin, H, W, Cout, pad, Kp, act);
            else
                conv_wmma_k<1,1><<<g, 128, 0, stream>>>(in, wptr[l], bias, out,
                    resid, 8, Cin, H, W, Cout, pad, Kp, act);
        }
    };

    auto bn = [&](float* t, int Cc, int HW, const float* g, const float* b, int relu) {
        bn_stats_k<<<Cc, 256, 0, stream>>>(t, STATS, 8, Cc, HW);
        size_t tot = (size_t)8 * Cc * HW;
        int blocks = (int)((tot + 255) / 256); if (blocks > 16384) blocks = 16384;
        bn_apply_k<<<blocks, 256, 0, stream>>>(t, STATS, g, b, Cc, HW, tot, relu);
    };
    auto fin = [&](int i) { return (const float*)d_in[i]; };

    // ------------------------- forward pipeline ---------------------------
    conv(x, 0, fin(2), A, nullptr, 256, 256, 0);                     // conv1
    maxpool2_k<<<2048, 256, 0, stream>>>(A, Bb, 8 * 16, 256, 256, 0);
    bn(Bb, 16, 16384, fin(3), fin(4), 1);                            // bn1 + relu
    conv(Bb, 1, fin(6), C, nullptr, 128, 128, 0);                    // conv2
    maxpool2_k<<<2048, 256, 0, stream>>>(C, D, 8 * 100, 128, 128, 1); // pool + relu
    conv(D, 2, fin(8),  E, nullptr, 64, 64, 0);                      // conv3
    conv(E, 3, fin(10), F, nullptr, 64, 64, 0);                      // conv4
    conv(F, 4, fin(12), G, nullptr, 64, 64, 1);                      // conv5 + relu
    resize9_k<<<3, 256, 0, stream>>>(G, Hh);
    concat_blur_k<<<8192, 256, 0, stream>>>(Hh, C, I);               // [8,181,128,128]

    conv(I, 5, fin(14), J, nullptr, 128, 128, 0);                    // res1_conv1
    bn(J, 256, 16384, fin(15), fin(16), 1);
    conv(J, 6, fin(18), K, nullptr, 128, 128, 0);                    // res1_conv2
    bn(K, 32, 16384, fin(19), fin(20), 0);
    conv(I, 7, fin(22), L, K, 128, 128, 0);                          // conv1x1_1 + add

    conv(L, 8, fin(24), M, nullptr, 128, 128, 0);                    // res2_conv1
    bn(M, 128, 16384, fin(25), fin(26), 1);
    conv(M, 9, fin(28), N, nullptr, 128, 128, 0);                    // res2_conv2
    bn(N, 64, 16384, fin(29), fin(30), 0);
    conv(L, 10, fin(32), O, N, 128, 128, 0);                         // conv1x1_2 + add

    conv(O, 11, fin(34), P, nullptr, 128, 128, 1);                   // up_conv1 + relu
    conv(P, 12, fin(36), Q, nullptr, 128, 128, 0);                   // up_conv2
    pixel_shuffle2_k<<<8192, 256, 0, stream>>>(Q, R);
    conv(R, 13, fin(38), S, nullptr, 256, 256, 1);                   // up_conv3 + relu
    conv(S, 14, fin(40), (float*)d_out, nullptr, 256, 256, 2);       // up_conv4 + sigmoid
}
